// LSTM_20074677141565
// MI455X (gfx1250) — matrix operand order repaired and checked by reading, not compiled, not run on hardware
//
#include <hip/hip_runtime.h>

typedef __attribute__((ext_vector_type(4)))  float  v4f;
typedef __attribute__((ext_vector_type(8)))  float  v8f;
typedef __attribute__((ext_vector_type(4)))  __bf16 v4bf;
typedef __attribute__((ext_vector_type(8)))  __bf16 v8bf;
typedef __attribute__((ext_vector_type(16))) __bf16 v16bf;

#define B_DIM  8192
#define K_IN   2048
#define H_DIM  2048
#define MT 128               // workgroup M tile
#define NT 64                // workgroup N tile (per gate)
#define KT 32                // K slab (one wmma_16x16x32 deep)
#define KP 40                // padded LDS K-stride in bf16 (80B, 16B-aligned)
#define NSLAB ((K_IN + H_DIM) / KT)   // 128

__device__ __forceinline__ float sigf(float x) {
    return 1.0f / (1.0f + __expf(-x));
}
__device__ __forceinline__ float tanh_fast(float x) {
    return 2.0f * sigf(2.0f * x) - 1.0f;   // tanh(x) = 2*sigmoid(2x) - 1
}

__global__ __launch_bounds__(256)
void lstm_wmma_kernel(const float* __restrict__ in_x,
                      const float* __restrict__ in_h,
                      const float* __restrict__ c_prev,
                      const float* __restrict__ Wii, const float* __restrict__ Wif,
                      const float* __restrict__ Wig, const float* __restrict__ Wio,
                      const float* __restrict__ Whi, const float* __restrict__ Whf,
                      const float* __restrict__ Whg, const float* __restrict__ Who,
                      const float* __restrict__ b_i, const float* __restrict__ b_f,
                      const float* __restrict__ b_g, const float* __restrict__ b_o,
                      float* __restrict__ out)
{
    // Double-buffered bf16 staging tiles.
    __shared__ __bf16 lsA[2][MT * KP];        // A tile, row-major [m][k]
    __shared__ __bf16 lsB[2][4][NT * KP];     // W tiles, transposed [n][k], per gate

    const int tid  = threadIdx.x;
    const int lane = tid & 31;
    const int wv   = tid >> 5;      // wave 0..7
    const int wm   = wv >> 2;       // 0..1 : wave M position (64-row slabs)
    const int wn   = wv & 3;        // 0..3 : wave N position (16-col slabs)
    const int half = lane >> 4;     // 0/1  : half-wave (fragment layout)
    const int l16  = lane & 15;

    const int mWG = blockIdx.y * MT;
    const int nWG = blockIdx.x * NT;

    const float* Wi[4] = {Wii, Wif, Wig, Wio};
    const float* Wh[4] = {Whi, Whf, Whg, Who};

    // A staging: thread handles rows (tid>>3)+32*i, 4 floats at col (tid&7)*4.
    const int rA = tid >> 3;
    const int cA = (tid & 7) * 4;
    // W staging (transpose-in-registers): lane n = tid&31 (+32*h), k-chunk (tid>>5)*4.
    const int nW = tid & 31;
    const int kW = (tid >> 5) * 4;

    v8f acc[4][4];                  // [gate][m-subtile]
    #pragma unroll
    for (int g = 0; g < 4; ++g)
        #pragma unroll
        for (int s = 0; s < 4; ++s)
            acc[g][s] = (v8f){0.f, 0.f, 0.f, 0.f, 0.f, 0.f, 0.f, 0.f};

    v4f   ra[4];                    // A staging regs
    float rw[4][2][4];              // W staging regs [gate][n-half][k-in-chunk]

    // Prologue: global-load slab 0 (input phase).
    #pragma unroll
    for (int i = 0; i < 4; ++i)
        ra[i] = *(const v4f*)(in_x + (size_t)(mWG + rA + 32 * i) * K_IN + cA);
    #pragma unroll
    for (int g = 0; g < 4; ++g)
        #pragma unroll
        for (int h = 0; h < 2; ++h)
            #pragma unroll
            for (int j = 0; j < 4; ++j)
                rw[g][h][j] = Wi[g][(size_t)(kW + j) * H_DIM + nWG + nW + 32 * h];

    int cur = 0;
    for (int ks = 0; ks < NSLAB; ++ks) {
        // ---- stage registers -> LDS (fp32 -> bf16), all b64 stores ----
        #pragma unroll
        for (int i = 0; i < 4; ++i) {
            v4bf p;
            p.x = (__bf16)ra[i].x; p.y = (__bf16)ra[i].y;
            p.z = (__bf16)ra[i].z; p.w = (__bf16)ra[i].w;
            *(v4bf*)(&lsA[cur][(rA + 32 * i) * KP + cA]) = p;
        }
        #pragma unroll
        for (int g = 0; g < 4; ++g)
            #pragma unroll
            for (int h = 0; h < 2; ++h) {
                v4bf p;
                p.x = (__bf16)rw[g][h][0]; p.y = (__bf16)rw[g][h][1];
                p.z = (__bf16)rw[g][h][2]; p.w = (__bf16)rw[g][h][3];
                *(v4bf*)(&lsB[cur][g][(nW + 32 * h) * KP + kW]) = p;
            }

        // ---- prefetch next slab into registers (overlaps with compute) ----
        if (ks + 1 < NSLAB) {
            const int  kk  = (ks + 1) * KT;
            const bool ph0 = (kk < K_IN);
            const int  kb  = ph0 ? kk : (kk - K_IN);
            const float* Ap = ph0 ? in_x : in_h;
            #pragma unroll
            for (int i = 0; i < 4; ++i)
                ra[i] = *(const v4f*)(Ap + (size_t)(mWG + rA + 32 * i) * K_IN + kb + cA);
            #pragma unroll
            for (int g = 0; g < 4; ++g) {
                const float* Wg = ph0 ? Wi[g] : Wh[g];
                #pragma unroll
                for (int h = 0; h < 2; ++h)
                    #pragma unroll
                    for (int j = 0; j < 4; ++j)
                        rw[g][h][j] = Wg[(size_t)(kb + kW + j) * H_DIM + nWG + nW + 32 * h];
            }
        }

        __syncthreads();

        // ---- fragment loads + WMMA ----
        // A 16x32 bf16: lane<16: M=lane, K chunks {0..7}@+0, {16..23}@+16
        //               lane>=16: M=lane-16, chunks {8..15}@+8, {24..31}@+24
        v16bf afr[4];
        #pragma unroll
        for (int s = 0; s < 4; ++s) {
            const __bf16* p = &lsA[cur][(wm * 64 + s * 16 + l16) * KP + (half ? 8 : 0)];
            v8bf lo = *(const v8bf*)(p);
            v8bf hi = *(const v8bf*)(p + 16);
            afr[s] = __builtin_shufflevector(lo, hi,
                     0,1,2,3,4,5,6,7,8,9,10,11,12,13,14,15);
        }
        // B 32x16 bf16: lane<16: N=lane, K {0..7},{16..23} ; lane>=16: N=lane-16, K {8..15},{24..31}
        #pragma unroll
        for (int g = 0; g < 4; ++g) {
            const __bf16* q = &lsB[cur][g][(wn * 16 + l16) * KP + (half ? 8 : 0)];
            v8bf lo = *(const v8bf*)(q);
            v8bf hi = *(const v8bf*)(q + 16);
            v16bf bfr = __builtin_shufflevector(lo, hi,
                        0,1,2,3,4,5,6,7,8,9,10,11,12,13,14,15);
            #pragma unroll
            for (int s = 0; s < 4; ++s)
                acc[g][s] = __builtin_amdgcn_wmma_f32_16x16x32_bf16(
                    false, afr[s], false, bfr, (short)0, acc[g][s], false, false);
        }

        cur ^= 1;
    }

    // ---- fused LSTM epilogue ----
    const int nLane = nWG + wn * 16 + l16;
    float bia[4];
    bia[0] = b_i[nLane]; bia[1] = b_f[nLane];
    bia[2] = b_g[nLane]; bia[3] = b_o[nLane];

    const size_t BH = (size_t)B_DIM * H_DIM;
    #pragma unroll
    for (int s = 0; s < 4; ++s) {
        const int mBase = mWG + wm * 64 + s * 16 + half * 8;  // C/D: M = r + 8*half
        #pragma unroll
        for (int r = 0; r < 8; ++r) {
            const size_t idx = (size_t)(mBase + r) * H_DIM + nLane;
            const float pi = acc[0][s][r] + bia[0];
            const float pf = acc[1][s][r] + bia[1];
            const float pg = acc[2][s][r] + bia[2];
            const float po = acc[3][s][r] + bia[3];
            const float it = sigf(pi);
            const float ft = sigf(pf);
            const float gt = tanh_fast(pg);
            const float ot = sigf(po);
            const float c  = ft * c_prev[idx] + it * gt;
            const float h  = ot * tanh_fast(c);
            out[idx]      = c;   // c_t
            out[BH + idx] = h;   // h_t
        }
    }
}

extern "C" void kernel_launch(void* const* d_in, const int* in_sizes, int n_in,
                              void* d_out, int out_size, void* d_ws, size_t ws_size,
                              hipStream_t stream) {
    (void)in_sizes; (void)n_in; (void)out_size; (void)d_ws; (void)ws_size;
    dim3 grid(H_DIM / NT, B_DIM / MT);   // (32, 64)
    lstm_wmma_kernel<<<grid, dim3(256), 0, stream>>>(
        (const float*)d_in[0],  (const float*)d_in[1],  (const float*)d_in[2],
        (const float*)d_in[3],  (const float*)d_in[4],  (const float*)d_in[5],
        (const float*)d_in[6],  (const float*)d_in[7],  (const float*)d_in[8],
        (const float*)d_in[9],  (const float*)d_in[10], (const float*)d_in[11],
        (const float*)d_in[12], (const float*)d_in[13], (const float*)d_in[14],
        (float*)d_out);
}
